// selfAttention_10325101379823
// MI455X (gfx1250) — matrix operand order repaired and checked by reading, not compiled, not run on hardware
//
#include <hip/hip_runtime.h>
#include <hip/hip_bf16.h>

// ---------------------------------------------------------------------------
// Non-local self-attention block for MI455X (gfx1250, wave32, WMMA).
// All GEMMs run on v_wmma_f32_16x16x32_f16 (f16 inputs, f32 accumulate) with
// register blocking so each fragment load feeds multiple WMMAs.
// Async global->LDS (ASYNCcnt) used for tile staging in the attention kernel.
// ---------------------------------------------------------------------------

#define BB 32
#define CC 512
#define NN 1024
#define CI 256

typedef __attribute__((ext_vector_type(16))) _Float16 v16h;
typedef __attribute__((ext_vector_type(8)))  _Float16 v8h;
typedef __attribute__((ext_vector_type(8)))  float    v8f;

// ---- WMMA fragment helpers (ISA 7.12.2 layouts, wave32) -------------------
// A (16x32 f16): lane holds row m=lane&15; elems e<8 -> K=8g+e, e>=8 -> K=16+8g+(e-8)
__device__ __forceinline__ v16h loadA_h(const _Float16* __restrict__ base, int ld, int lane) {
    const int m = lane & 15, g = lane >> 4;
    const _Float16* p = base + (size_t)m * ld + 8 * g;
    v8h lo = *(const v8h*)(p);
    v8h hi = *(const v8h*)(p + 16);
    v16h a;
#pragma unroll
    for (int e = 0; e < 8; ++e) { a[e] = lo[e]; a[8 + e] = hi[e]; }
    return a;
}

// B (32x16 f16) from a K-contiguous (transposed) store: lane holds col n=lane&15;
// elem e<8 -> K = 8g + e, e>=8 -> K = 16 + 8g + (e-8)  (matches A layout)
__device__ __forceinline__ v16h loadB_h(const _Float16* __restrict__ baseT, int ld, int lane) {
    const int n = lane & 15, g = lane >> 4;
    const _Float16* p = baseT + (size_t)n * ld + 8 * g;
    v8h lo = *(const v8h*)(p);
    v8h hi = *(const v8h*)(p + 16);
    v16h b;
#pragma unroll
    for (int e = 0; e < 8; ++e) { b[e] = lo[e]; b[8 + e] = hi[e]; }
    return b;
}

__device__ __forceinline__ v8f wmma16(v16h a, v16h b, v8f c) {
    return __builtin_amdgcn_wmma_f32_16x16x32_f16(false, a, false, b, (short)0, c, false, false);
}

// ---------------------------------------------------------------------------
// Kernel 0: convert weights to f16.
//   WzH   [C][Ci]           (K-contiguous B frags for GEMM3)
//   WallH [3*Ci][C]         rows 0..255 = Wt, 256..511 = Wp, 512..767 = Wg
// ---------------------------------------------------------------------------
__global__ void k_cvt(const float* __restrict__ Wz, const float* __restrict__ Wt,
                      const float* __restrict__ Wp, const float* __restrict__ Wg,
                      _Float16* __restrict__ WzH, _Float16* __restrict__ WallH) {
    const int SZ = CC * CI;                       // 131072 (all four are this size)
    int i = blockIdx.x * 256 + threadIdx.x;       // 0 .. 4*SZ-1
    if (i < SZ) { WzH[i] = (_Float16)Wz[i]; return; }
    i -= SZ;
    const int sel = i / SZ, off = i % SZ;
    const float* src = (sel == 0) ? Wt : ((sel == 1) ? Wp : Wg);
    WallH[i] = (_Float16)src[off];
}

// ---------------------------------------------------------------------------
// Kernel 1: projections  out[o][n] = sum_c W[o][c] * x[b][c][n] + bias[o]
// One block per (batch, 16-column panel): stage x panel once into LDS as
// transposed f16, then 8 waves sweep all 48 o-tiles (6 each). The shared B
// fragment is hoisted: per k-step, 1 B load pair feeds 6 WMMAs.
//   theta -> thetaH [b][n][Ci], phi -> phiH [b][n][Ci], g -> gH [b][Ci][N]
// ---------------------------------------------------------------------------
#define XROW 520   // 512 + 8 f16 pad: B-frag lanes land on distinct LDS banks

__global__ void k_proj(const float* __restrict__ x,
                       const _Float16* __restrict__ WallH,
                       const float* __restrict__ bt, const float* __restrict__ bp,
                       const float* __restrict__ bg,
                       _Float16* __restrict__ thetaH, _Float16* __restrict__ phiH,
                       _Float16* __restrict__ gH) {
    __shared__ _Float16 xT[16 * XROW];            // 16.6 KB
    const int tid = threadIdx.x;
    const int lane = tid & 31, wave = tid >> 5;
    const int b  = blockIdx.x >> 6;
    const int nt = blockIdx.x & 63;
    const int n0 = nt * 16;

    // stage x[b][:, n0:n0+16] -> xT[n][c] f16 (coalesced 64B row reads)
    for (int c = tid; c < CC; c += 256) {
        const float4* p = (const float4*)(x + ((size_t)b * CC + c) * NN + n0);
        const float4 v0 = p[0], v1 = p[1], v2 = p[2], v3 = p[3];
        _Float16* q = xT + c;
        q[0 * XROW] = (_Float16)v0.x;  q[1 * XROW] = (_Float16)v0.y;
        q[2 * XROW] = (_Float16)v0.z;  q[3 * XROW] = (_Float16)v0.w;
        q[4 * XROW] = (_Float16)v1.x;  q[5 * XROW] = (_Float16)v1.y;
        q[6 * XROW] = (_Float16)v1.z;  q[7 * XROW] = (_Float16)v1.w;
        q[8 * XROW] = (_Float16)v2.x;  q[9 * XROW] = (_Float16)v2.y;
        q[10 * XROW] = (_Float16)v2.z; q[11 * XROW] = (_Float16)v2.w;
        q[12 * XROW] = (_Float16)v3.x; q[13 * XROW] = (_Float16)v3.y;
        q[14 * XROW] = (_Float16)v3.z; q[15 * XROW] = (_Float16)v3.w;
    }
    __syncthreads();

    const int m = lane & 15, g = lane >> 4;
    const int ncol = n0 + m;

    v8f acc[6] = {};
#pragma unroll 2
    for (int k0 = 0; k0 < CC; k0 += 32) {
        const v16h bb = loadB_h(xT + k0, XROW, lane);      // shared by 6 tiles
#pragma unroll
        for (int i = 0; i < 6; ++i) {
            const int ot = wave * 6 + i;
            v16h a = loadA_h(WallH + (size_t)(ot * 16) * CC + k0, CC, lane);
            acc[i] = wmma16(a, bb, acc[i]);
        }
    }

#pragma unroll
    for (int i = 0; i < 6; ++i) {
        const int ot = wave * 6 + i;                       // 0..47
        const int proj = ot >> 4;                          // 0=theta 1=phi 2=g
        const int o0 = (ot & 15) * 16;
        const float* bias = (proj == 0) ? bt : ((proj == 1) ? bp : bg);
        if (proj != 2) {
            // [b][n][o]: o = o0+8g+r contiguous -> one 16B store
            _Float16* dst = ((proj == 0) ? thetaH : phiH) +
                            ((size_t)b * NN + ncol) * CI + o0 + 8 * g;
            v8h outv;
#pragma unroll
            for (int r = 0; r < 8; ++r) outv[r] = (_Float16)(acc[i][r] + bias[o0 + 8 * g + r]);
            *(v8h*)dst = outv;
        } else {
            // gH [b][o][n]
#pragma unroll
            for (int r = 0; r < 8; ++r) {
                const int oo = o0 + 8 * g + r;
                gH[((size_t)b * CI + oo) * NN + ncol] = (_Float16)(acc[i][r] + bias[oo]);
            }
        }
    }
}

// ---------------------------------------------------------------------------
// Kernel 2: fused attention per (batch, 32-row tile).
//   S = theta_tile * phi^T   (K=Ci)          -> LDS f32 [32][1032]
//   softmax rows             (8 lanes/row)   -> LDS f16 [32][1040]
//   y = attn * g             (K=N)           -> LDS f16 [32][264]
//   w_y = y * Wz^T + bz      (K=Ci)          -> global f32 [b][c][n] (pre-BN)
// Register-blocked: 2x4 (GEMM1), 2x2 (GEMM2), 2x4 (GEMM3).
// ---------------------------------------------------------------------------
#define SROW 1032
#define AROW 1040
#define YROW 264
#define SMEM_BYTES (32*CI*2 + 32*SROW*4 + 32*AROW*2 + 32*YROW*2)

__global__ void k_attn(const _Float16* __restrict__ thetaH,
                       const _Float16* __restrict__ phiH,
                       const _Float16* __restrict__ gH,
                       const _Float16* __restrict__ WzH,
                       const float* __restrict__ bz,
                       float* __restrict__ wy) {
    extern __shared__ char smem[];
    _Float16* thL = (_Float16*)smem;                                   // 16 KB @ LDS off 0
    float*    S   = (float*)(smem + 32 * CI * 2);                      // 129 KB
    _Float16* At  = (_Float16*)(smem + 32 * CI * 2 + 32 * SROW * 4);   // 65 KB
    _Float16* yL  = (_Float16*)((char*)At + 32 * AROW * 2);            // 16.5 KB

    const int tid  = threadIdx.x;
    const int lane = tid & 31, wave = tid >> 5;
    const int b    = blockIdx.x >> 5;
    const int rt   = blockIdx.x & 31;
    const int row0 = rt * 32;
    const int nloc = lane & 15, g = lane >> 4;

    // ---- stage theta rows [row0,row0+32) x Ci into LDS: async DMA, 16 KB
    {
        const unsigned long long gbase =
            (unsigned long long)(const void*)(thetaH + ((size_t)b * NN + row0) * CI);
#pragma unroll
        for (int i = 0; i < 4; ++i) {
            const unsigned ldsOff = (unsigned)(tid + i * 256) * 16u;   // thL at LDS offset 0
            const unsigned long long ga = gbase + ldsOff;
            asm volatile("global_load_async_to_lds_b128 %0, %1, off"
                         :: "v"(ldsOff), "v"(ga) : "memory");
        }
        asm volatile("s_wait_asynccnt 0" ::: "memory");
    }
    __syncthreads();

    // ---- GEMM1: scores S[32][1024]; per wave: 2 groups of a 2(m)x4(n) block
    const _Float16* phiB = phiH + (size_t)b * NN * CI;   // [m][Ci]
    for (int grp = 0; grp < 2; ++grp) {
        const int ntbase = wave * 8 + grp * 4;
        v8f acc0[4] = {}, acc1[4] = {};
#pragma unroll 2
        for (int k0 = 0; k0 < CI; k0 += 32) {
            const v16h a0 = loadA_h(thL + k0, CI, lane);
            const v16h a1 = loadA_h(thL + 16 * CI + k0, CI, lane);
#pragma unroll
            for (int j = 0; j < 4; ++j) {
                const v16h bb = loadB_h(phiB + (size_t)((ntbase + j) * 16) * CI + k0, CI, lane);
                acc0[j] = wmma16(a0, bb, acc0[j]);
                acc1[j] = wmma16(a1, bb, acc1[j]);
            }
        }
#pragma unroll
        for (int j = 0; j < 4; ++j) {
            const int col = (ntbase + j) * 16 + nloc;
#pragma unroll
            for (int r = 0; r < 8; ++r) {
                S[(8 * g + r) * SROW + col]      = acc0[j][r];
                S[(16 + 8 * g + r) * SROW + col] = acc1[j][r];
            }
        }
    }
    __syncthreads();

    // ---- softmax: 8 lanes per row, shuffle reductions (wave32)
    {
        const int row = tid >> 3, sub = tid & 7;
        float* sr = S + row * SROW;
        float mx = -3.402823e38f;
        for (int j = sub; j < NN; j += 8) mx = fmaxf(mx, sr[j]);
        mx = fmaxf(mx, __shfl_xor(mx, 1));
        mx = fmaxf(mx, __shfl_xor(mx, 2));
        mx = fmaxf(mx, __shfl_xor(mx, 4));
        float sum = 0.f;
        for (int j = sub; j < NN; j += 8) { float e = __expf(sr[j] - mx); sr[j] = e; sum += e; }
        sum += __shfl_xor(sum, 1);
        sum += __shfl_xor(sum, 2);
        sum += __shfl_xor(sum, 4);
        const float inv = 1.0f / sum;
        _Float16* ar = At + row * AROW;
        for (int j = sub; j < NN; j += 8) ar[j] = (_Float16)(sr[j] * inv);
    }
    __syncthreads();

    // ---- GEMM2: y[32][Ci] = attn * g; full 2(m)x2(o) block per wave, K=N
    const _Float16* gB = gH + (size_t)b * CI * NN;       // [o][m]
    {
        const int ot0 = wave * 2;
        v8f acc00 = {}, acc01 = {}, acc10 = {}, acc11 = {};
#pragma unroll 2
        for (int k0 = 0; k0 < NN; k0 += 32) {
            __builtin_prefetch(gB + (size_t)(ot0 * 16) * NN + k0 + 128, 0, 0);
            const v16h a0 = loadA_h(At + k0, AROW, lane);
            const v16h a1 = loadA_h(At + 16 * AROW + k0, AROW, lane);
            const v16h b0 = loadB_h(gB + (size_t)(ot0 * 16) * NN + k0, NN, lane);
            const v16h b1 = loadB_h(gB + (size_t)((ot0 + 1) * 16) * NN + k0, NN, lane);
            acc00 = wmma16(a0, b0, acc00);
            acc10 = wmma16(a1, b0, acc10);
            acc01 = wmma16(a0, b1, acc01);
            acc11 = wmma16(a1, b1, acc11);
        }
        const int col0 = ot0 * 16 + nloc, col1 = (ot0 + 1) * 16 + nloc;
#pragma unroll
        for (int r = 0; r < 8; ++r) {
            yL[(8 * g + r) * YROW + col0]      = (_Float16)acc00[r];
            yL[(16 + 8 * g + r) * YROW + col0] = (_Float16)acc10[r];
            yL[(8 * g + r) * YROW + col1]      = (_Float16)acc01[r];
            yL[(16 + 8 * g + r) * YROW + col1] = (_Float16)acc11[r];
        }
    }
    __syncthreads();

    // ---- GEMM3: w_y[32][C] = y * Wz^T + bz; 2(m)x4(c) block per wave, K=Ci
    {
        const int ctbase = wave * 4;
        v8f acc0[4] = {}, acc1[4] = {};
#pragma unroll 2
        for (int k0 = 0; k0 < CI; k0 += 32) {
            const v16h a0 = loadA_h(yL + k0, YROW, lane);
            const v16h a1 = loadA_h(yL + 16 * YROW + k0, YROW, lane);
#pragma unroll
            for (int j = 0; j < 4; ++j) {
                const v16h bb = loadB_h(WzH + (size_t)((ctbase + j) * 16) * CI + k0, CI, lane);
                acc0[j] = wmma16(a0, bb, acc0[j]);
                acc1[j] = wmma16(a1, bb, acc1[j]);
            }
        }
#pragma unroll
        for (int j = 0; j < 4; ++j) {
            const int c = (ctbase + j) * 16 + nloc;
            const float bias = bz[c];
            float* dst0 = wy + ((size_t)b * CC + c) * NN + row0 + 8 * g;
            float* dst1 = dst0 + 16;
#pragma unroll
            for (int r = 0; r < 8; ++r) {
                dst0[r] = acc0[j][r] + bias;    // rows contiguous in n
                dst1[r] = acc1[j][r] + bias;
            }
        }
    }
}

// ---------------------------------------------------------------------------
// Kernel 3: per-channel batch-norm statistics over (B, N)
// ---------------------------------------------------------------------------
__global__ void k_stats(const float* __restrict__ wy,
                        float* __restrict__ mean, float* __restrict__ rstd) {
    __shared__ float s1[256], s2[256];
    const int c = blockIdx.x, tid = threadIdx.x;
    float s = 0.f, ss = 0.f;
    for (int b = 0; b < BB; ++b) {
        const float4 v = ((const float4*)(wy + ((size_t)b * CC + c) * NN))[tid];
        s  += v.x + v.y + v.z + v.w;
        ss += v.x * v.x + v.y * v.y + v.z * v.z + v.w * v.w;
    }
    s1[tid] = s; s2[tid] = ss;
    __syncthreads();
    for (int off = 128; off; off >>= 1) {
        if (tid < off) { s1[tid] += s1[tid + off]; s2[tid] += s2[tid + off]; }
        __syncthreads();
    }
    if (tid == 0) {
        const float inv = 1.0f / (float)(BB * NN);
        const float m = s1[0] * inv;
        const float var = s2[0] * inv - m * m;
        mean[c] = m;
        rstd[c] = rsqrtf(var + 1e-5f);
    }
}

// ---------------------------------------------------------------------------
// Kernel 4: BN affine + residual, float4 vectorized
// ---------------------------------------------------------------------------
__global__ void k_apply(const float* __restrict__ wy, const float* __restrict__ x,
                        const float* __restrict__ mean, const float* __restrict__ rstd,
                        const float* __restrict__ gamma, const float* __restrict__ beta,
                        float* __restrict__ out) {
    const size_t i4 = (size_t)blockIdx.x * 256 + threadIdx.x;
    const size_t total4 = (size_t)BB * CC * NN / 4;
    if (i4 >= total4) return;
    const int c = (int)((i4 * 4 / NN) % CC);
    const float m = mean[c], r = rstd[c], ga = gamma[c], be = beta[c];
    const float4 v  = ((const float4*)wy)[i4];
    const float4 xi = ((const float4*)x)[i4];
    float4 o;
    o.x = (v.x - m) * r * ga + be + xi.x;
    o.y = (v.y - m) * r * ga + be + xi.y;
    o.z = (v.z - m) * r * ga + be + xi.z;
    o.w = (v.w - m) * r * ga + be + xi.w;
    ((float4*)out)[i4] = o;
}

// ---------------------------------------------------------------------------
extern "C" void kernel_launch(void* const* d_in, const int* in_sizes, int n_in,
                              void* d_out, int out_size, void* d_ws, size_t ws_size,
                              hipStream_t stream) {
    const float* x     = (const float*)d_in[0];
    const float* Wg    = (const float*)d_in[1];
    const float* bg    = (const float*)d_in[2];
    const float* Wt    = (const float*)d_in[3];
    const float* bt    = (const float*)d_in[4];
    const float* Wp    = (const float*)d_in[5];
    const float* bp    = (const float*)d_in[6];
    const float* Wz    = (const float*)d_in[7];
    const float* bz    = (const float*)d_in[8];
    const float* gamma = (const float*)d_in[9];
    const float* beta  = (const float*)d_in[10];
    float* out = (float*)d_out;

    // workspace layout (bytes)
    char* ws = (char*)d_ws;
    const size_t szProj = (size_t)BB * NN * CI * sizeof(_Float16);   // 16 MB each
    const size_t szW    = (size_t)CC * CI * sizeof(_Float16);        // 256 KB
    _Float16* thetaH = (_Float16*)(ws);
    _Float16* phiH   = (_Float16*)(ws + szProj);
    _Float16* gH     = (_Float16*)(ws + 2 * szProj);
    _Float16* WzH    = (_Float16*)(ws + 3 * szProj);
    _Float16* WallH  = (_Float16*)(ws + 3 * szProj + szW);
    float*    wyBuf  = (float*)   (ws + 3 * szProj + 4 * szW);
    float*    meanB  = (float*)((char*)wyBuf + (size_t)BB * CC * NN * sizeof(float));
    float*    rstdB  = meanB + CC;

    // weights -> f16 (Wz + packed Wt/Wp/Wg panel)
    k_cvt<<<(4 * CC * CI) / 256, 256, 0, stream>>>(Wz, Wt, Wp, Wg, WzH, WallH);

    // projections: one block per (batch, 16-col panel)
    k_proj<<<BB * 64, 256, 0, stream>>>(x, WallH, bt, bp, bg, thetaH, phiH, gH);

    // fused attention (score GEMM + softmax + attn*g + Wz GEMM)
    k_attn<<<BB * 32, 256, SMEM_BYTES, stream>>>(thetaH, phiH, gH, WzH, bz, wyBuf);

    k_stats<<<CC, 256, 0, stream>>>(wyBuf, meanB, rstdB);

    k_apply<<<(BB * CC * NN / 4 + 255) / 256, 256, 0, stream>>>(
        wyBuf, x, meanB, rstdB, gamma, beta, out);
}
